// ViT_26130581028874
// MI455X (gfx1250) — matrix operand order repaired and checked by reading, not compiled, not run on hardware
//
#include <hip/hip_runtime.h>
#include <hip/hip_bf16.h>

typedef __attribute__((ext_vector_type(16))) _Float16 v16h;
typedef __attribute__((ext_vector_type(8)))  _Float16 v8h;
typedef __attribute__((ext_vector_type(8)))  float    v8f;
typedef __attribute__((ext_vector_type(4)))  int      v4i;

#define ACT_NONE 0
#define ACT_RELU 1
#define ACT_GELU 2

// ---------------------------------------------------------------------------
// WMMA helpers (layouts per CDNA5 ISA §7.12.2, wave32)
// ---------------------------------------------------------------------------
__device__ __forceinline__ v8f wmma16(v16h a, v16h b, v8f c) {
  return __builtin_amdgcn_wmma_f32_16x16x32_f16(false, a, false, b, (short)0, c,
                                                false, false);
}

__device__ __forceinline__ v16h pack16(v8h lo, v8h hi) {
  v16h a;
#pragma unroll
  for (int i = 0; i < 8; ++i) { a[i] = lo[i]; a[i + 8] = hi[i]; }
  return a;
}

// A fragment: row-major A[M,K], lane l -> row m0+(l&15),
// chunks at k = kb+(l>>4)*8 and kb+16+(l>>4)*8 (each 8 contiguous halves).
__device__ __forceinline__ v16h load_a_frag(const _Float16* A, long lda, int m0,
                                            int kb, int lane) {
  int r = lane & 15, hf = lane >> 4;
  const _Float16* p = A + (long)(m0 + r) * lda + kb + hf * 8;
  v8h lo = *(const v8h*)p;
  v8h hi = *(const v8h*)(p + 16);
  return pack16(lo, hi);
}

// B fragment from global: weights stored [N,K] row-major; lane l -> col
// n0+(l&15), chunks at k = kb+(l>>4)*8 and kb+16+(l>>4)*8 (each 8 contiguous halves).
__device__ __forceinline__ v16h load_b_frag(const _Float16* Bw, long ldb, int n0,
                                            int kb, int lane) {
  int c = lane & 15, hf = lane >> 4;
  const _Float16* p = Bw + (long)(n0 + c) * ldb + kb + hf * 8;
  v8h lo = *(const v8h*)p;
  v8h hi = *(const v8h*)(p + 16);
  return pack16(lo, hi);
}

__device__ __forceinline__ float activate(float v, int act) {
  if (act == ACT_RELU) return fmaxf(v, 0.0f);
  if (act == ACT_GELU) return 0.5f * v * (1.0f + erff(v * 0.70710678118654752f));
  return v;
}

// ---------------------------------------------------------------------------
// CDNA5 async staging: GLOBAL_LOAD_ASYNC_TO_LDS_B128 (ASYNCcnt) with fallback.
// Builtin signature (from compiler diagnostics): param0 = AS1 v4i* (global
// source), param1 = LDS v4i* (dest), then offset + cpol.
// ---------------------------------------------------------------------------
#if __has_builtin(__builtin_amdgcn_global_load_async_to_lds_b128)
#define ATHENA_ASYNC_LDS 1
#else
#define ATHENA_ASYNC_LDS 0
#endif

typedef __attribute__((address_space(1))) v4i gv4i_t;
typedef __attribute__((address_space(3))) v4i lv4i_t;

__device__ __forceinline__ void stage16(const _Float16* src, _Float16* dst_lds) {
#if ATHENA_ASYNC_LDS
  __builtin_amdgcn_global_load_async_to_lds_b128((gv4i_t*)src, (lv4i_t*)dst_lds,
                                                 0, 0);
#else
  *(v8h*)dst_lds = *(const v8h*)src;
#endif
}

__device__ __forceinline__ void wait_stage() {
#if ATHENA_ASYNC_LDS
#if __has_builtin(__builtin_amdgcn_s_wait_asynccnt)
  __builtin_amdgcn_s_wait_asynccnt(0);
#else
  asm volatile("s_wait_asynccnt 0x0" ::: "memory");
#endif
#endif
}

// ---------------------------------------------------------------------------
// LDS-staged WMMA GEMM: C = A[M,K] * W[N,K]^T, fused epilogue.
// Block = 8 waves = 8 M-tiles (128 rows) sharing one 64-wide N-block.
// Per 64-deep K chunk the block async-stages a 64x64 B tile into LDS
// (8 KB, padded rows to dodge bank conflicts); each wave then runs
// 2 k-steps x 4 WMMAs reading B fragments from LDS.
// Grid: (M/128, N/64). Requires M%128==0, N%64==0, K%64==0.
// ---------------------------------------------------------------------------
#define BROW 72  // 64 halves data + 8 halves pad (144 B row stride)

__global__ void gemm_f16_lds(const _Float16* __restrict__ A, long lda,
                             const _Float16* __restrict__ Bw, long ldb,
                             int M, int N, int K,
                             const float* __restrict__ scale,
                             const float* __restrict__ bias,
                             const _Float16* __restrict__ residH, long ldrh,
                             const float* __restrict__ residF, long ldrf,
                             int act,
                             _Float16* __restrict__ outH, long ldoh,
                             float* __restrict__ outF, long ldof, int nvalid) {
  __shared__ _Float16 Bs[64 * BROW];
  int tid = threadIdx.x;
  int lane = tid & 31, wv = tid >> 5;
  int m0 = (blockIdx.x * 8 + wv) * 16;
  int nb = blockIdx.y * 64;
  (void)M; (void)N;

  v8f acc[4];
#pragma unroll
  for (int nn = 0; nn < 4; ++nn)
#pragma unroll
    for (int i = 0; i < 8; ++i) acc[nn][i] = 0.0f;

  int r = lane & 15, hf = lane >> 4;

  for (int kc = 0; kc < K; kc += 64) {
    __syncthreads();  // previous chunk fully consumed
    {
      // 64 rows x 64 halves = 512 x 16B transfers; 2 per thread
      int t0 = tid, t1 = tid + 256;
      int row0 = t0 >> 3, seg0 = t0 & 7;
      int row1 = t1 >> 3, seg1 = t1 & 7;
      stage16(Bw + (long)(nb + row0) * ldb + kc + seg0 * 8,
              &Bs[row0 * BROW + seg0 * 8]);
      stage16(Bw + (long)(nb + row1) * ldb + kc + seg1 * 8,
              &Bs[row1 * BROW + seg1 * 8]);
    }
    if (kc + 64 < K)  // prefetch next A chunk (global_prefetch_b8)
      __builtin_prefetch(A + (long)(m0 + r) * lda + kc + 64, 0, 3);
    wait_stage();
    __syncthreads();

#pragma unroll
    for (int kk = 0; kk < 64; kk += 32) {
      v16h a = load_a_frag(A, lda, m0, kc + kk, lane);
#pragma unroll
      for (int nn = 0; nn < 4; ++nn) {
        const _Float16* p = &Bs[(nn * 16 + r) * BROW + kk + hf * 8];
        v16h bf = pack16(*(const v8h*)p, *(const v8h*)(p + 16));
        acc[nn] = wmma16(a, bf, acc[nn]);
      }
    }
  }

#pragma unroll
  for (int nn = 0; nn < 4; ++nn) {
#pragma unroll
    for (int i = 0; i < 8; ++i) {
      int row = m0 + i + hf * 8;
      int col = nb + nn * 16 + r;
      float v = acc[nn][i];
      if (scale)  v *= scale[col];
      if (bias)   v += bias[col];
      if (residH) v += (float)residH[(long)row * ldrh + col];
      if (residF) v += residF[(long)row * ldrf + col];
      v = activate(v, act);
      if (outH) outH[(long)row * ldoh + col] = (_Float16)v;
      if (outF && col < nvalid) outF[(long)row * ldof + col] = v;
    }
  }
}

// ---------------------------------------------------------------------------
// Direct-load WMMA GEMM (for small M, e.g. the classification head).
// One wave computes a 16 x (16*NT) tile.
// ---------------------------------------------------------------------------
template <int NT>
__global__ void gemm_f16(const _Float16* __restrict__ A, long lda,
                         const _Float16* __restrict__ Bw, long ldb,
                         int M, int N, int K,
                         const float* __restrict__ scale,
                         const float* __restrict__ bias,
                         int act,
                         _Float16* __restrict__ outH, long ldoh,
                         float* __restrict__ outF, long ldof, int nvalid) {
  int lane = threadIdx.x & 31;
  int wave = blockIdx.x * (blockDim.x >> 5) + (threadIdx.x >> 5);
  int tiles_n = N / (16 * NT);
  int tiles = (M / 16) * tiles_n;
  if (wave >= tiles) return;
  int tm = wave / tiles_n, tn = wave % tiles_n;
  int m0 = tm * 16, nb = tn * 16 * NT;

  v8f acc[NT];
#pragma unroll
  for (int nn = 0; nn < NT; ++nn)
#pragma unroll
    for (int i = 0; i < 8; ++i) acc[nn][i] = 0.0f;

  for (int kb = 0; kb < K; kb += 32) {
    v16h a = load_a_frag(A, lda, m0, kb, lane);
#pragma unroll
    for (int nn = 0; nn < NT; ++nn) {
      v16h b = load_b_frag(Bw, ldb, nb + nn * 16, kb, lane);
      acc[nn] = wmma16(a, b, acc[nn]);
    }
  }

  int r = lane & 15, hf = lane >> 4;
#pragma unroll
  for (int nn = 0; nn < NT; ++nn) {
#pragma unroll
    for (int i = 0; i < 8; ++i) {
      int row = m0 + i + hf * 8;
      int col = nb + nn * 16 + r;
      float v = acc[nn][i];
      if (scale) v *= scale[col];
      if (bias)  v += bias[col];
      v = activate(v, act);
      if (outH) outH[(long)row * ldoh + col] = (_Float16)v;
      if (outF && col < nvalid) outF[(long)row * ldof + col] = v;
    }
  }
}

// ---------------------------------------------------------------------------
// 3x3 conv 64->64 pad=1 as 9-tap implicit WMMA GEMM over NHWC-f16 activations.
// Wt layout [9][64][64] = [tap][N][K]. Epilogue: BN + ReLU.
// ---------------------------------------------------------------------------
__global__ void conv3x3_wmma(const _Float16* __restrict__ in,
                             const _Float16* __restrict__ Wt,
                             const float* __restrict__ scale,
                             const float* __restrict__ bias,
                             _Float16* __restrict__ out) {
  int lane = threadIdx.x & 31;
  int wave = blockIdx.x * (blockDim.x >> 5) + (threadIdx.x >> 5);
  if (wave >= (32 * 4096) / 16) return;
  int m0 = wave * 16;
  int b = m0 >> 12, rem = m0 & 4095;
  int y = rem >> 6, x0 = rem & 63;  // tile stays within one image row (16 | 64)
  int r = lane & 15, hf = lane >> 4;
  int x = x0 + r;

  v8f acc[4];
#pragma unroll
  for (int nn = 0; nn < 4; ++nn)
#pragma unroll
    for (int i = 0; i < 8; ++i) acc[nn][i] = 0.0f;

#pragma unroll
  for (int tap = 0; tap < 9; ++tap) {
    int dy = tap / 3 - 1, dx = tap % 3 - 1;
    int yy = y + dy, xx = x + dx;
    bool valid = (yy >= 0) && (yy < 64) && (xx >= 0) && (xx < 64);
    const _Float16* ap = in + (((long)b * 64 + yy) * 64 + xx) * 64;
    const _Float16* wp = Wt + tap * 4096;
#pragma unroll
    for (int kb = 0; kb < 64; kb += 32) {
      v8h lo, hi;
      if (valid) {
        lo = *(const v8h*)(ap + kb + hf * 8);
        hi = *(const v8h*)(ap + kb + 16 + hf * 8);
      } else {
#pragma unroll
        for (int i = 0; i < 8; ++i) { lo[i] = (_Float16)0.0f; hi[i] = (_Float16)0.0f; }
      }
      v16h a = pack16(lo, hi);
#pragma unroll
      for (int nn = 0; nn < 4; ++nn) {
        v16h bf = load_b_frag(wp, 64, nn * 16, kb, lane);
        acc[nn] = wmma16(a, bf, acc[nn]);
      }
    }
  }

#pragma unroll
  for (int nn = 0; nn < 4; ++nn) {
#pragma unroll
    for (int i = 0; i < 8; ++i) {
      int pix = m0 + i + hf * 8;
      int co = nn * 16 + r;
      float v = acc[nn][i] * scale[co] + bias[co];
      v = fmaxf(v, 0.0f);
      out[(long)pix * 64 + co] = (_Float16)v;
    }
  }
}

// ---------------------------------------------------------------------------
// Patch-embedding GEMM: A gathered from NHWC-f16 [32,64,64,256] per
//   b c (h p1)(w p2) -> b (h w) (p1 p2 c); f = (p1*4+p2)*256 + c.
// K = 4096, N = 512. Epilogue: + patch_b + pos, store fp32 into X[b,1+tok,:].
// ---------------------------------------------------------------------------
__device__ __forceinline__ v8h patch_chunk(const _Float16* act, int b, int hh,
                                           int ww, int k) {
  int p12 = k >> 8, c = k & 255;
  int p1 = p12 >> 2, p2 = p12 & 3;
  const _Float16* p =
      act + (((long)b * 64 + hh * 4 + p1) * 64 + (ww * 4 + p2)) * 256 + c;
  return *(const v8h*)p;
}

__global__ void gemm_patch(const _Float16* __restrict__ act,
                           const _Float16* __restrict__ Bw,
                           const float* __restrict__ bias,
                           const float* __restrict__ pos,
                           float* __restrict__ X) {
  int lane = threadIdx.x & 31;
  int wave = blockIdx.x * (blockDim.x >> 5) + (threadIdx.x >> 5);
  const int tiles_n = 512 / 64;
  int tiles = (8192 / 16) * tiles_n;
  if (wave >= tiles) return;
  int tm = wave / tiles_n, tn = wave % tiles_n;
  int m0 = tm * 16, nb = tn * 64;
  int r = lane & 15, hf = lane >> 4;
  int trow = m0 + r;
  int b = trow >> 8, tok = trow & 255;
  int hh = tok >> 4, ww = tok & 15;

  v8f acc[4];
#pragma unroll
  for (int nn = 0; nn < 4; ++nn)
#pragma unroll
    for (int i = 0; i < 8; ++i) acc[nn][i] = 0.0f;

  for (int kb = 0; kb < 4096; kb += 32) {
    v8h lo = patch_chunk(act, b, hh, ww, kb + hf * 8);
    v8h hi = patch_chunk(act, b, hh, ww, kb + 16 + hf * 8);
    v16h a = pack16(lo, hi);
#pragma unroll
    for (int nn = 0; nn < 4; ++nn) {
      v16h bf = load_b_frag(Bw, 4096, nb + nn * 16, kb, lane);
      acc[nn] = wmma16(a, bf, acc[nn]);
    }
  }

#pragma unroll
  for (int nn = 0; nn < 4; ++nn) {
#pragma unroll
    for (int i = 0; i < 8; ++i) {
      int row = m0 + i + hf * 8;
      int b2 = row >> 8, tok2 = row & 255;
      int col = nb + nn * 16 + r;
      float v = acc[nn][i] + bias[col] + pos[(long)(1 + tok2) * 512 + col];
      X[((long)b2 * 288 + 1 + tok2) * 512 + col] = v;
    }
  }
}

// ---------------------------------------------------------------------------
// LayerNorm over 512 features: one wave per row, shfl reductions.
// ---------------------------------------------------------------------------
__global__ void ln_kernel(const float* __restrict__ X, _Float16* __restrict__ Y,
                          const float* __restrict__ g, const float* __restrict__ bb,
                          int rows, long xstride, long ystride) {
  int wave = blockIdx.x * (blockDim.x >> 5) + (threadIdx.x >> 5);
  if (wave >= rows) return;
  int lane = threadIdx.x & 31;
  const float* x = X + (long)wave * xstride;
  float s = 0.0f, ss = 0.0f;
  for (int i = lane; i < 512; i += 32) {
    float v = x[i];
    s += v; ss += v * v;
  }
#pragma unroll
  for (int m = 16; m >= 1; m >>= 1) {
    s  += __shfl_xor(s,  m, 32);
    ss += __shfl_xor(ss, m, 32);
  }
  float mean = s * (1.0f / 512.0f);
  float var = ss * (1.0f / 512.0f) - mean * mean;
  float inv = rsqrtf(var + 1e-5f);
  _Float16* y = Y + (long)wave * ystride;
  for (int i = lane; i < 512; i += 32)
    y[i] = (_Float16)(((x[i] - mean) * inv) * g[i] + bb[i]);
}

// ---------------------------------------------------------------------------
// Attention: one block per (token i, b*8+h). 257 tokens, head_dim 64.
// scale = DIM^-0.5 (full dim, per reference).
// ---------------------------------------------------------------------------
__global__ void attn_kernel(const _Float16* __restrict__ QKV,
                            _Float16* __restrict__ AO) {
  __shared__ float qsh[64];
  __shared__ float sc[260];
  __shared__ float red[256];
  const float SCALE = 0.04419417382415922f;  // 512^-0.5
  int tid = threadIdx.x;
  int i = blockIdx.x;        // 0..256
  int bh = blockIdx.y;       // b*8+h
  int b = bh >> 3, h = bh & 7;
  const _Float16* base = QKV + (long)b * 288 * 1536;
  const _Float16* qp = base + (long)i * 1536 + h * 64;
  if (tid < 64) qsh[tid] = (float)qp[tid];
  __syncthreads();

  for (int j = tid; j < 257; j += 256) {
    const _Float16* kp = base + (long)j * 1536 + 512 + h * 64;
    float d = 0.0f;
#pragma unroll 8
    for (int dd = 0; dd < 64; ++dd) d += qsh[dd] * (float)kp[dd];
    sc[j] = d * SCALE;
  }
  __syncthreads();

  float m = -1e30f;
  for (int j = tid; j < 257; j += 256) m = fmaxf(m, sc[j]);
  red[tid] = m;
  __syncthreads();
  for (int s = 128; s > 0; s >>= 1) {
    if (tid < s) red[tid] = fmaxf(red[tid], red[tid + s]);
    __syncthreads();
  }
  float mx = red[0];
  __syncthreads();

  float ls = 0.0f;
  for (int j = tid; j < 257; j += 256) {
    float e = expf(sc[j] - mx);
    sc[j] = e;
    ls += e;
  }
  red[tid] = ls;
  __syncthreads();
  for (int s = 128; s > 0; s >>= 1) {
    if (tid < s) red[tid] += red[tid + s];
    __syncthreads();
  }
  float inv = 1.0f / red[0];
  __syncthreads();

  if (tid < 64) {
    const _Float16* vp = base + 1024 + h * 64 + tid;
    float acc = 0.0f;
    for (int j = 0; j < 257; ++j) acc += sc[j] * (float)vp[(long)j * 1536];
    AO[((long)b * 288 + i) * 512 + h * 64 + tid] = (_Float16)(acc * inv);
  }
}

// ---------------------------------------------------------------------------
// Stem conv1 3x3 3->64 pad=1 + BN + ReLU, NCHW fp32 in -> NHWC f16 out.
// ---------------------------------------------------------------------------
__global__ void conv1_kernel(const float* __restrict__ img,
                             const float* __restrict__ w,
                             const float* __restrict__ scale,
                             const float* __restrict__ bias,
                             _Float16* __restrict__ out) {
  long idx = (long)blockIdx.x * blockDim.x + threadIdx.x;
  if (idx >= 32L * 4096 * 64) return;
  int co = (int)(idx & 63);
  long p = idx >> 6;
  int x = (int)(p & 63), y = (int)((p >> 6) & 63), b = (int)(p >> 12);
  float acc = 0.0f;
#pragma unroll
  for (int ci = 0; ci < 3; ++ci)
#pragma unroll
    for (int ky = 0; ky < 3; ++ky)
#pragma unroll
      for (int kx = 0; kx < 3; ++kx) {
        int yy = y + ky - 1, xx = x + kx - 1;
        if (yy >= 0 && yy < 64 && xx >= 0 && xx < 64)
          acc += img[(((long)b * 3 + ci) * 64 + yy) * 64 + xx] *
                 w[((co * 3 + ci) * 3 + ky) * 3 + kx];
      }
  float v = fmaxf(acc * scale[co] + bias[co], 0.0f);
  out[p * 64 + co] = (_Float16)v;
}

// ---------------------------------------------------------------------------
// Prep kernels
// ---------------------------------------------------------------------------
__global__ void bn_prep(const float* g, const float* b, const float* m,
                        const float* v, float* scale, float* bias, int C) {
  int i = threadIdx.x + blockIdx.x * blockDim.x;
  if (i < C) {
    float s = g[i] * rsqrtf(v[i] + 1e-5f);
    scale[i] = s;
    bias[i] = b[i] - m[i] * s;
  }
}

__global__ void cvt_f16(const float* __restrict__ src, _Float16* __restrict__ dst,
                        long n) {
  long i = (long)blockIdx.x * blockDim.x + threadIdx.x;
  if (i < n) dst[i] = (_Float16)src[i];
}

__global__ void repack3(const float* __restrict__ w, _Float16* __restrict__ out) {
  int idx = blockIdx.x * blockDim.x + threadIdx.x;
  if (idx >= 9 * 4096) return;
  int tap = idx >> 12, nk = idx & 4095;
  int n = nk >> 6, k = nk & 63;
  int dy = tap / 3, dx = tap % 3;
  out[idx] = (_Float16)w[(((long)n * 64 + k) * 3 + dy) * 3 + dx];
}

__global__ void pad_w2(const float* __restrict__ w, _Float16* __restrict__ out) {
  int idx = blockIdx.x * blockDim.x + threadIdx.x;
  if (idx >= 112 * 512) return;
  int n = idx >> 9, k = idx & 511;
  out[idx] = (_Float16)(n < 100 ? w[(long)n * 512 + k] : 0.0f);
}

__global__ void pad_b2(const float* __restrict__ b, float* __restrict__ out) {
  int i = threadIdx.x;
  if (i < 112) out[i] = (i < 100) ? b[i] : 0.0f;
}

__global__ void setup_x(const float* __restrict__ cls, const float* __restrict__ pos,
                        float* __restrict__ X) {
  long idx = (long)blockIdx.x * blockDim.x + threadIdx.x;
  if (idx >= 32L * 288 * 512) return;
  int c = (int)(idx & 511);
  long t = idx >> 9;
  int tok = (int)(t % 288);
  float v = 0.0f;
  if (tok == 0) v = cls[c] + pos[c];
  X[idx] = v;  // tokens 1..256 overwritten by gemm_patch; 257..287 stay 0
}

__global__ void zero_ao_pad(_Float16* __restrict__ AO) {
  long idx = (long)blockIdx.x * blockDim.x + threadIdx.x;
  if (idx >= 32L * 31 * 512) return;
  int c = (int)(idx & 511);
  long r = idx >> 9;
  int b = (int)(r / 31), tok = 257 + (int)(r % 31);
  AO[((long)b * 288 + tok) * 512 + c] = (_Float16)0.0f;
}

// ---------------------------------------------------------------------------
// Host launcher
// ---------------------------------------------------------------------------
extern "C" void kernel_launch(void* const* d_in, const int* in_sizes, int n_in,
                              void* d_out, int out_size, void* d_ws, size_t ws_size,
                              hipStream_t stream) {
  (void)in_sizes; (void)n_in; (void)out_size; (void)ws_size;
  const float* P[132];
  for (int i = 0; i < 132; ++i) P[i] = (const float*)d_in[i];
  // Input index map (depth-first, dict insertion order):
  // 0 img | 1 conv1 | 2-5 bn1 g/b/m/v
  // block1: 6 c1, 7-10 bn1, 11 c2, 12-15 bn2, 16 c3, 17-20 bn3, 21 sc, 22-25 scbn
  // block2: 26 c1, 27-30 bn1, 31 c2, 32-35 bn2, 36 c3, 37-40 bn3
  // block3: 41 c1, 42-45 bn1, 46 c2, 47-50 bn2, 51 c3, 52-55 bn3
  // 56 patch_w, 57 patch_b, 58 cls, 59 pos
  // layer l: base=60+11l: ln1.g,ln1.b,qkv,out_w,out_b,ln2.g,ln2.b,w1,b1,w2,b2
  // 126-127 head_ln g/b, 128 head_w1, 129 head_b1, 130 head_w2, 131 head_b2

  char* base = (char*)d_ws;
  size_t off = 0;
  auto wsa = [&](size_t bytes) -> void* {
    void* p = base + off;
    off = (off + bytes + 255) & ~(size_t)255;
    return p;
  };

  const long NPIX = 32L * 4096;   // 131072 pixels
  const long MTOK = 32L * 288;    // 9216 padded token rows

  // activations
  _Float16* X64   = (_Float16*)wsa(NPIX * 64 * 2);
  _Float16* T64A  = (_Float16*)wsa(NPIX * 64 * 2);
  _Float16* T64B  = (_Float16*)wsa(NPIX * 64 * 2);
  _Float16* X256A = (_Float16*)wsa(NPIX * 256 * 2);
  _Float16* X256B = (_Float16*)wsa(NPIX * 256 * 2);
  _Float16* SC256 = (_Float16*)wsa(NPIX * 256 * 2);
  float*    Xtok  = (float*)wsa(MTOK * 512 * 4);
  _Float16* Yln   = (_Float16*)wsa(MTOK * 512 * 2);
  _Float16* QKVb  = (_Float16*)wsa(MTOK * 1536 * 2);
  _Float16* AObuf = (_Float16*)wsa(MTOK * 512 * 2);
  _Float16* Hbuf  = (_Float16*)wsa(MTOK * 512 * 2);
  _Float16* CLSY  = (_Float16*)wsa(32L * 512 * 2);
  _Float16* H1    = (_Float16*)wsa(32L * 512 * 2);

  struct BN { float* s; float* b; };
  auto prep_bn = [&](int gi, int C) -> BN {
    BN r;
    r.s = (float*)wsa((size_t)C * 4);
    r.b = (float*)wsa((size_t)C * 4);
    bn_prep<<<1, 256, 0, stream>>>(P[gi], P[gi + 1], P[gi + 2], P[gi + 3], r.s, r.b, C);
    return r;
  };
  auto cvt = [&](const float* src, long n) -> _Float16* {
    _Float16* d = (_Float16*)wsa((size_t)n * 2);
    cvt_f16<<<(int)((n + 255) / 256), 256, 0, stream>>>(src, d, n);
    return d;
  };
  auto repack = [&](const float* src) -> _Float16* {
    _Float16* d = (_Float16*)wsa((size_t)9 * 4096 * 2);
    repack3<<<(9 * 4096 + 255) / 256, 256, 0, stream>>>(src, d);
    return d;
  };

  // ---- weight prep ----
  BN bnTop = prep_bn(2, 64);
  BN b1n1 = prep_bn(7, 64),  b1n2 = prep_bn(12, 64), b1n3 = prep_bn(17, 256),
     b1sc = prep_bn(22, 256);
  BN b2n1 = prep_bn(27, 64), b2n2 = prep_bn(32, 64), b2n3 = prep_bn(37, 256);
  BN b3n1 = prep_bn(42, 64), b3n2 = prep_bn(47, 64), b3n3 = prep_bn(52, 256);

  _Float16* w_b1c1 = cvt(P[6], 64 * 64);
  _Float16* w_b1t2 = repack(P[11]);
  _Float16* w_b1c3 = cvt(P[16], 256 * 64);
  _Float16* w_b1sc = cvt(P[21], 256 * 64);
  _Float16* w_b2c1 = cvt(P[26], 64 * 256);
  _Float16* w_b2t2 = repack(P[31]);
  _Float16* w_b2c3 = cvt(P[36], 256 * 64);
  _Float16* w_b3c1 = cvt(P[41], 64 * 256);
  _Float16* w_b3t2 = repack(P[46]);
  _Float16* w_b3c3 = cvt(P[51], 256 * 64);
  _Float16* patchW = cvt(P[56], 512L * 4096);
  _Float16 *qkv16[6], *out16[6], *w116[6], *w216[6];
  for (int l = 0; l < 6; ++l) {
    int b0 = 60 + 11 * l;
    qkv16[l] = cvt(P[b0 + 2], 1536L * 512);
    out16[l] = cvt(P[b0 + 3], 512L * 512);
    w116[l]  = cvt(P[b0 + 7], 512L * 512);
    w216[l]  = cvt(P[b0 + 9], 512L * 512);
  }
  _Float16* hw1 = cvt(P[128], 512L * 512);
  _Float16* hw2 = (_Float16*)wsa(112L * 512 * 2);
  pad_w2<<<(112 * 512 + 255) / 256, 256, 0, stream>>>(P[130], hw2);
  float* hb2 = (float*)wsa(112 * 4);
  pad_b2<<<1, 128, 0, stream>>>(P[131], hb2);

  // LDS-staged GEMM launcher: grid (M/128, N/64), 256 threads.
  auto g4 = [&](const _Float16* A, long lda, const _Float16* Bw, long ldb,
                int M, int N, int K, const float* sc, const float* bi,
                const _Float16* rh, long ldrh, const float* rf, long ldrf,
                int act, _Float16* oh, long ldoh, float* of, long ldof, int nv) {
    dim3 grid(M / 128, N / 64);
    gemm_f16_lds<<<grid, 256, 0, stream>>>(A, lda, Bw, ldb, M, N, K, sc, bi, rh,
                                           ldrh, rf, ldrf, act, oh, ldoh, of,
                                           ldof, nv);
  };

  // ---- stem ----
  conv1_kernel<<<(int)((32L * 4096 * 64 + 255) / 256), 256, 0, stream>>>(
      P[0], P[1], bnTop.s, bnTop.b, X64);

  // ---- bottleneck 1 (shortcut conv) ----
  g4(X64, 64, w_b1sc, 64, (int)NPIX, 256, 64, b1sc.s, b1sc.b, nullptr, 0,
     nullptr, 0, ACT_NONE, SC256, 256, nullptr, 0, 0);
  g4(X64, 64, w_b1c1, 64, (int)NPIX, 64, 64, b1n1.s, b1n1.b, nullptr, 0,
     nullptr, 0, ACT_RELU, T64A, 64, nullptr, 0, 0);
  conv3x3_wmma<<<1024, 256, 0, stream>>>(T64A, w_b1t2, b1n2.s, b1n2.b, T64B);
  g4(T64B, 64, w_b1c3, 64, (int)NPIX, 256, 64, b1n3.s, b1n3.b, SC256, 256,
     nullptr, 0, ACT_RELU, X256A, 256, nullptr, 0, 0);

  // ---- bottleneck 2 (identity shortcut) ----
  g4(X256A, 256, w_b2c1, 256, (int)NPIX, 64, 256, b2n1.s, b2n1.b, nullptr, 0,
     nullptr, 0, ACT_RELU, T64A, 64, nullptr, 0, 0);
  conv3x3_wmma<<<1024, 256, 0, stream>>>(T64A, w_b2t2, b2n2.s, b2n2.b, T64B);
  g4(T64B, 64, w_b2c3, 64, (int)NPIX, 256, 64, b2n3.s, b2n3.b, X256A, 256,
     nullptr, 0, ACT_RELU, X256B, 256, nullptr, 0, 0);

  // ---- bottleneck 3 ----
  g4(X256B, 256, w_b3c1, 256, (int)NPIX, 64, 256, b3n1.s, b3n1.b, nullptr, 0,
     nullptr, 0, ACT_RELU, T64A, 64, nullptr, 0, 0);
  conv3x3_wmma<<<1024, 256, 0, stream>>>(T64A, w_b3t2, b3n2.s, b3n2.b, T64B);
  g4(T64B, 64, w_b3c3, 64, (int)NPIX, 256, 64, b3n3.s, b3n3.b, X256B, 256,
     nullptr, 0, ACT_RELU, X256A, 256, nullptr, 0, 0);

  // ---- patch embed + cls + pos ----
  setup_x<<<(int)((32L * 288 * 512 + 255) / 256), 256, 0, stream>>>(P[58], P[59], Xtok);
  gemm_patch<<<512, 256, 0, stream>>>(X256A, patchW, P[57], P[59], Xtok);
  zero_ao_pad<<<(int)((32L * 31 * 512 + 255) / 256), 256, 0, stream>>>(AObuf);

  // ---- transformer layers ----
  for (int l = 0; l < 6; ++l) {
    int b0 = 60 + 11 * l;
    ln_kernel<<<(int)((MTOK + 7) / 8), 256, 0, stream>>>(Xtok, Yln, P[b0], P[b0 + 1],
                                                         (int)MTOK, 512, 512);
    g4(Yln, 512, qkv16[l], 512, (int)MTOK, 1536, 512, nullptr, nullptr, nullptr,
       0, nullptr, 0, ACT_NONE, QKVb, 1536, nullptr, 0, 0);
    attn_kernel<<<dim3(257, 256), 256, 0, stream>>>(QKVb, AObuf);
    g4(AObuf, 512, out16[l], 512, (int)MTOK, 512, 512, nullptr, P[b0 + 4],
       nullptr, 0, Xtok, 512, ACT_NONE, nullptr, 0, Xtok, 512, 1 << 30);
    ln_kernel<<<(int)((MTOK + 7) / 8), 256, 0, stream>>>(Xtok, Yln, P[b0 + 5],
                                                         P[b0 + 6], (int)MTOK, 512, 512);
    g4(Yln, 512, w116[l], 512, (int)MTOK, 512, 512, nullptr, P[b0 + 8], nullptr,
       0, nullptr, 0, ACT_GELU, Hbuf, 512, nullptr, 0, 0);
    g4(Hbuf, 512, w216[l], 512, (int)MTOK, 512, 512, nullptr, P[b0 + 10],
       nullptr, 0, Xtok, 512, ACT_NONE, nullptr, 0, Xtok, 512, 1 << 30);
  }

  // ---- head (small M: direct-load WMMA GEMM) ----
  ln_kernel<<<(32 + 7) / 8, 256, 0, stream>>>(Xtok, CLSY, P[126], P[127], 32,
                                              288L * 512, 512);
  {
    int tiles1 = (32 / 16) * (512 / 64);  // 16 waves
    gemm_f16<4><<<(tiles1 + 7) / 8, 256, 0, stream>>>(
        CLSY, 512, hw1, 512, 32, 512, 512, nullptr, P[129], ACT_GELU, H1, 512,
        nullptr, 0, 0);
    int tiles2 = (32 / 16) * (112 / 16);  // 14 waves
    gemm_f16<1><<<(tiles2 + 7) / 8, 256, 0, stream>>>(
        H1, 512, hw2, 512, 32, 112, 512, nullptr, hb2, ACT_NONE, nullptr, 0,
        (float*)d_out, 100, 100);
  }
}
